// CrossAttention_63960652972380
// MI455X (gfx1250) — compile-verified, matrix operand order checked by reading, not run on hardware
//
#include <hip/hip_runtime.h>
#include <hip/hip_bf16.h>
#include <math.h>

typedef __attribute__((ext_vector_type(16))) _Float16 v16h;
typedef __attribute__((ext_vector_type(2)))  __fp16   v2fp16;
typedef __attribute__((ext_vector_type(8)))  float    v8f;

#define B_   16
#define L_   4096
#define CH_  512
#define NH_  8
#define HD_  64
#define KK_  256
#define TL_  128   // rows per workgroup tile

// ---- fragment K-pair map (ISA 7.12.2, 16-bit A 16x32): VGPR r holds K pair ----
__device__ __forceinline__ int kpair(int r, int hi) {
  return ((r < 4) ? (2 * r) : (2 * r + 8)) + (hi ? 8 : 0);
}

// Load an A fragment (16x32 f16) from LDS, row-major with stride `strideH` halves.
__device__ __forceinline__ v16h load_a(const _Float16* base, int row, int strideH,
                                       int kbase, int hi) {
  union { v16h v; unsigned int u[8]; } x;
#pragma unroll
  for (int r = 0; r < 8; ++r) {
    int k = kbase + kpair(r, hi);                 // even -> aligned 32-bit pair
    x.u[r] = *(const unsigned int*)(base + row * strideH + k);
  }
  return x.v;
}

__device__ __forceinline__ v8f wmma_f16(v16h a, v16h b, v8f c) {
  return __builtin_amdgcn_wmma_f32_16x16x32_f16(false, a, false, b, (short)0, c,
                                                false, false);
}

// Load 4 B fragments (one per owned n-tile) for k-step `ks`.
__device__ __forceinline__ void load_b4(const _Float16* Bsw, int w, int lane, int ks,
                                        v16h bf[4]) {
#pragma unroll
  for (int nt = 0; nt < 4; ++nt)
    bf[nt] = *(const v16h*)(Bsw + (size_t)(((4 * w + nt) * 16 + ks) * 32 + lane) * 16);
}

// One k-step of the 128x64x512 per-wave GEMM; A double-buffered across row-stripes.
__device__ __forceinline__ void gemm_kstep(const _Float16* Albs, const v16h bf[4],
                                           int ks, int l15, int hi, v8f acc[8][4]) {
  v16h af = load_a(Albs, l15, CH_, ks * 32, hi);
#pragma unroll
  for (int rs = 0; rs < 8; ++rs) {
    v16h ac = af;
    if (rs < 7) af = load_a(Albs, (rs + 1) * 16 + l15, CH_, ks * 32, hi);
#pragma unroll
    for (int nt = 0; nt < 4; ++nt)
      acc[rs][nt] = wmma_f16(ac, bf[nt], acc[rs][nt]);
  }
}

// Full K=512 accumulation with ping-pong B buffers (static buffer indices).
__device__ __forceinline__ void gemm_tile(const _Float16* Albs, const _Float16* Bsw,
                                          int w, int lane, int l15, int hi,
                                          v8f acc[8][4]) {
#pragma unroll
  for (int rs = 0; rs < 8; ++rs)
#pragma unroll
    for (int nt = 0; nt < 4; ++nt) { v8f z = {}; acc[rs][nt] = z; }
  v16h b0[4], b1[4];
  load_b4(Bsw, w, lane, 0, b0);
#pragma unroll 1
  for (int ks0 = 0; ks0 < 16; ks0 += 2) {
    load_b4(Bsw, w, lane, ks0 + 1, b1);
    if (ks0 + 2 < 16)
      __builtin_prefetch(Bsw + (size_t)((4 * w) * 16 + ks0 + 2) * 512 + lane * 16, 0, 1);
    gemm_kstep(Albs, b0, ks0, l15, hi, acc);
    if (ks0 + 2 < 16) load_b4(Bsw, w, lane, ks0 + 2, b0);
    gemm_kstep(Albs, b1, ks0 + 1, l15, hi, acc);
  }
}

// ============================ prep: weight swizzle ============================
// B-fragment layout: frag = nt*16 + ks ; chunk of 512 halves; lane owns 16 halves.
__global__ void prep_weights(const float* __restrict__ Wq, const float* __restrict__ Wo,
                             _Float16* __restrict__ wqS, _Float16* __restrict__ woS) {
  int t = blockIdx.x * 256 + threadIdx.x;          // 0..32767
  if (t >= 2 * 512 * 32) return;
  int lane = t & 31;
  int frag = (t >> 5) & 511;
  int mat  = t >> 14;
  int ks = frag & 15, nt = frag >> 4;
  int hi = (lane >= 16);
  int n  = nt * 16 + (lane & 15);
  const float* W = mat ? Wo : Wq;
  _Float16* dst  = (mat ? woS : wqS) + (size_t)frag * 512 + lane * 16;
#pragma unroll
  for (int r = 0; r < 8; ++r)
#pragma unroll
    for (int j = 0; j < 2; ++j) {
      int k = ks * 32 + kpair(r, hi) + j;
      dst[2 * r + j] = (_Float16)W[k * CH_ + n];
    }
}

// ===================== prep: k/v projection + swizzle =========================
// kS frag index: ((b*8+h)*16+kt)*2+dstep  (B = k^T : K-dim=d, N=key)
// vS frag index: ((b*8+h)*4+nt2)*8+ks2    (B = v   : K-dim=key, N=d)
__global__ void prep_kv(const float* __restrict__ ctx, const float* __restrict__ Wk,
                        const float* __restrict__ Wv, _Float16* __restrict__ kS,
                        _Float16* __restrict__ vS) {
  int t = blockIdx.x * 256 + threadIdx.x;          // 0..262143
  if (t >= 8192 * 32) return;
  int lane = t & 31, fragAll = t >> 5;
  int hi = (lane >= 16), l15 = lane & 15;
  if (fragAll < 4096) {                            // ---- kS ----
    int dstep = fragAll & 1, kt = (fragAll >> 1) & 15;
    int h = (fragAll >> 5) & 7, bb = fragAll >> 8;
    int key = kt * 16 + l15;
    const float* c = ctx + ((size_t)bb * KK_ + key) * 4;
    float c0 = c[0], c1 = c[1], c2 = c[2], c3 = c[3];
    _Float16* dst = kS + (size_t)fragAll * 512 + lane * 16;
#pragma unroll
    for (int r = 0; r < 8; ++r)
#pragma unroll
      for (int j = 0; j < 2; ++j) {
        int d = dstep * 32 + kpair(r, hi) + j;
        int col = h * HD_ + d;
        float val = c0 * Wk[col] + c1 * Wk[CH_ + col] +
                    c2 * Wk[2 * CH_ + col] + c3 * Wk[3 * CH_ + col];
        dst[2 * r + j] = (_Float16)val;
      }
  } else {                                         // ---- vS ----
    int g = fragAll - 4096;
    int ks2 = g & 7, nt2 = (g >> 3) & 3;
    int h = (g >> 5) & 7, bb = g >> 8;
    int d = nt2 * 16 + l15, col = h * HD_ + d;
    _Float16* dst = vS + (size_t)g * 512 + lane * 16;
#pragma unroll
    for (int r = 0; r < 8; ++r)
#pragma unroll
      for (int j = 0; j < 2; ++j) {
        int key = ks2 * 32 + kpair(r, hi) + j;
        const float* c = ctx + ((size_t)bb * KK_ + key) * 4;
        float val = c[0] * Wv[col] + c[1] * Wv[CH_ + col] +
                    c[2] * Wv[2 * CH_ + col] + c[3] * Wv[3 * CH_ + col];
        dst[2 * r + j] = (_Float16)val;
      }
  }
}

// ============================ fused main kernel ==============================
__global__ __launch_bounds__(256) void attn_main(
    const float* __restrict__ xF, const float* __restrict__ bo,
    const int* __restrict__ lengths, const _Float16* __restrict__ wqS,
    const _Float16* __restrict__ woS, const _Float16* __restrict__ kS,
    const _Float16* __restrict__ vS, float* __restrict__ out) {
  extern __shared__ _Float16 smem[];               // 256 KB dynamic LDS
  _Float16* xfh = smem;                            // [128][512] f16 (GEMM1 A)
  _Float16* qsm = smem + TL_ * CH_;                // [128][512] f16 (q, then O)
  _Float16* psm = smem;                            // reuse xfh: per-wave P [16][256]

  int bb   = blockIdx.x >> 5;
  int tile = blockIdx.x & 31;
  int l0   = tile * TL_;
  int tid  = threadIdx.x;
  int lane = tid & 31, w = tid >> 5;
  int hi = (lane >= 16), l15 = lane & 15;

  // ---- 1) stage xF tile into LDS as f16 (packed converts, 64-bit DS stores) ----
  const float* xtile = xF + ((size_t)(bb * L_ + l0)) * CH_;
  for (int i = tid; i < TL_ * CH_ / 4; i += 256) {
    float4 f = ((const float4*)xtile)[i];
    union { v2fp16 h; unsigned u; } p0, p1;
    p0.h = __builtin_amdgcn_cvt_pkrtz(f.x, f.y);
    p1.h = __builtin_amdgcn_cvt_pkrtz(f.z, f.w);
    uint2 st; st.x = p0.u; st.y = p1.u;
    ((uint2*)xfh)[i] = st;
  }
  __syncthreads();

  // ---- 2) GEMM1: q = xfh @ Wq ; wave w owns output n-tiles [4w,4w+4) ----
  {
    v8f acc[8][4];
    gemm_tile(xfh, wqS, w, lane, l15, hi, acc);
#pragma unroll
    for (int rs = 0; rs < 8; ++rs)
#pragma unroll
      for (int nt = 0; nt < 4; ++nt)
#pragma unroll
        for (int r = 0; r < 8; ++r)
          qsm[(rs * 16 + r + (hi ? 8 : 0)) * CH_ + (4 * w + nt) * 16 + l15] =
              (_Float16)acc[rs][nt][r];
  }
  __syncthreads();   // q visible to all waves; xfh region now dead -> P scratch

  // ---- 3) attention: wave w owns rows [16w,16w+16) of the tile ----
  const float scale = 0.125f;                       // 1/sqrt(64)
  _Float16* pw = psm + w * (16 * KK_);
#pragma unroll 1
  for (int hh = 0; hh < NH_; ++hh) {
    // A fragments of q for this head are kt-invariant: hoist (2 loads, not 32).
    v16h aq0 = load_a(qsm, w * 16 + l15, CH_, hh * HD_, hi);
    v16h aq1 = load_a(qsm, w * 16 + l15, CH_, hh * HD_ + 32, hi);
    v8f s[16];
#pragma unroll
    for (int kt = 0; kt < 16; ++kt) { v8f z = {}; s[kt] = z; }
#pragma unroll
    for (int kt = 0; kt < 16; ++kt) {
      const _Float16* p0 =
          kS + (size_t)((((bb * NH_ + hh) * 16 + kt) * 2 + 0) * 32 + lane) * 16;
      const _Float16* p1 =
          kS + (size_t)((((bb * NH_ + hh) * 16 + kt) * 2 + 1) * 32 + lane) * 16;
      s[kt] = wmma_f16(aq0, *(const v16h*)p0, s[kt]);
      s[kt] = wmma_f16(aq1, *(const v16h*)p1, s[kt]);
    }
    // softmax over 256 keys per row (row = r + 8*hi; keys across 16 lanes x 16 tiles)
#pragma unroll
    for (int r = 0; r < 8; ++r) {
      float m = -INFINITY;
#pragma unroll
      for (int kt = 0; kt < 16; ++kt) m = fmaxf(m, s[kt][r]);
#pragma unroll
      for (int off = 1; off < 16; off <<= 1) m = fmaxf(m, __shfl_xor(m, off, 32));
      float sum = 0.f;
#pragma unroll
      for (int kt = 0; kt < 16; ++kt) {
        float e = __expf((s[kt][r] - m) * scale);
        s[kt][r] = e;
        sum += e;
      }
#pragma unroll
      for (int off = 1; off < 16; off <<= 1) sum += __shfl_xor(sum, off, 32);
      float rinv = __builtin_amdgcn_rcpf(sum);
#pragma unroll
      for (int kt = 0; kt < 16; ++kt)
        pw[(r + (hi ? 8 : 0)) * KK_ + kt * 16 + l15] = (_Float16)(s[kt][r] * rinv);
    }
    // O = P @ V  -> overwrite consumed q columns of this head (own rows only)
    v8f oacc[4];
#pragma unroll
    for (int nt2 = 0; nt2 < 4; ++nt2) { v8f z = {}; oacc[nt2] = z; }
    v16h ap = load_a(pw, l15, KK_, 0, hi);
#pragma unroll 1
    for (int ks2 = 0; ks2 < 8; ++ks2) {
      v16h ac = ap;
      if (ks2 < 7) ap = load_a(pw, l15, KK_, (ks2 + 1) * 32, hi);
#pragma unroll
      for (int nt2 = 0; nt2 < 4; ++nt2) {
        const _Float16* p =
            vS + (size_t)((((bb * NH_ + hh) * 4 + nt2) * 8 + ks2) * 32 + lane) * 16;
        oacc[nt2] = wmma_f16(ac, *(const v16h*)p, oacc[nt2]);
      }
    }
#pragma unroll
    for (int nt2 = 0; nt2 < 4; ++nt2)
#pragma unroll
      for (int r = 0; r < 8; ++r)
        qsm[(w * 16 + r + (hi ? 8 : 0)) * CH_ + hh * HD_ + nt2 * 16 + l15] =
            (_Float16)oacc[nt2][r];
  }
  __syncthreads();   // O (in qsm) visible to all waves

  // ---- 4) GEMM3: out = O @ Wo + bo + xF, masked by lengths ----
  {
    v8f acc[8][4];
    gemm_tile(qsm, woS, w, lane, l15, hi, acc);
    int len = lengths[bb];
#pragma unroll
    for (int rs = 0; rs < 8; ++rs)
#pragma unroll
      for (int nt = 0; nt < 4; ++nt) {
        int col = 64 * w + nt * 16 + l15;
        float bv = bo[col];
#pragma unroll
        for (int r = 0; r < 8; ++r) {
          int row = rs * 16 + r + (hi ? 8 : 0);
          int l = l0 + row;
          size_t idx = ((size_t)(bb * L_ + l)) * CH_ + col;
          float res = acc[rs][nt][r] + bv + xF[idx];
          res = (l < len) ? res : 0.0f;
          __builtin_nontemporal_store(res, &out[idx]);   // out never re-read
        }
      }
  }
}

// ================================ launcher ===================================
extern "C" void kernel_launch(void* const* d_in, const int* in_sizes, int n_in,
                              void* d_out, int out_size, void* d_ws, size_t ws_size,
                              hipStream_t stream) {
  const float* xF   = (const float*)d_in[0];
  const float* ctx  = (const float*)d_in[1];
  const int*   len  = (const int*)d_in[2];
  const float* Wq   = (const float*)d_in[3];
  const float* Wk   = (const float*)d_in[4];
  const float* Wv   = (const float*)d_in[5];
  const float* Wo   = (const float*)d_in[6];
  const float* bo   = (const float*)d_in[7];
  float* out = (float*)d_out;

  _Float16* wsH = (_Float16*)d_ws;
  _Float16* wqS = wsH;                              // 512*512
  _Float16* woS = wqS + 512 * 512;                  // 512*512
  _Float16* kS  = woS + 512 * 512;                  // 4096 frags * 512
  _Float16* vS  = kS + 4096 * 512;                  // 4096 frags * 512

  prep_weights<<<128, 256, 0, stream>>>(Wq, Wo, wqS, woS);
  prep_kv<<<1024, 256, 0, stream>>>(ctx, Wk, Wv, kS, vS);

  size_t ldsBytes = (size_t)2 * TL_ * CH_ * sizeof(_Float16);   // 256 KB
  attn_main<<<B_ * (L_ / TL_), 256, ldsBytes, stream>>>(xF, bo, len, wqS, woS,
                                                        kS, vS, out);
}
